// GENModel_8323646619970
// MI455X (gfx1250) — matrix operand order repaired and checked by reading, not compiled, not run on hardware
//
#include <hip/hip_runtime.h>
#include <hip/hip_bf16.h>

// ---------------------------------------------------------------------------
// GENConv GNN pipeline for MI455X (gfx1250, wave32, WMMA).
//
// Dense linear layers -> v_wmma_f32_16x16x32_f16 (f32 data converted to f16
// in LDS, f32 accumulate). Segment softmax aggregation -> L2-resident atomics
// (node arrays <= 26MB << 192MB L2). Edge-wide [E,128] array is the HBM
// roofline term (~1.6GB traffic/layer @ 23.3TB/s).
// ---------------------------------------------------------------------------

typedef __attribute__((ext_vector_type(16))) _Float16 v16h;
typedef __attribute__((ext_vector_type(8)))  _Float16 v8h;
typedef __attribute__((ext_vector_type(8)))  float    v8f;

#define CAT16(lo, hi) __builtin_shufflevector((lo), (hi), 0,1,2,3,4,5,6,7,8,9,10,11,12,13,14,15)

#define N_NODES 50000
#define N_EDGES 800000
#define N_GRAPH 64
#define TILE_ROWS 128

// ------------------------------ WMMA GEMM ----------------------------------
// C[M,Ncol] = A[M,K] @ W[K,Ncol] + bias, optional fused ReLU.
// Block: 256 threads = 8 waves; each wave owns 16 rows of the 128-row tile.
// LDS: Wt (transposed f16 weights, [Ncol][Kpad+8]) + At (f16 A tile,
// [128][Kpad+8]) + bias. Row pad of 8 halfs (16B) kills bank conflicts
// while preserving 16B alignment for ds_load_b128.
__global__ void __launch_bounds__(256) wmma_gemm_bias(
    const float* __restrict__ A, const float* __restrict__ W,
    const float* __restrict__ bias, float* __restrict__ C,
    int M, int K, int Ncol, int fuse_relu)
{
    extern __shared__ _Float16 lds[];
    const int Kpad   = (K + 31) & ~31;
    const int lstr   = Kpad + 8;                 // halfs per LDS row
    _Float16* Wt = lds;                          // [Ncol][lstr]
    _Float16* At = lds + (size_t)Ncol * lstr;    // [TILE_ROWS][lstr]
    float*   bsh = (float*)(At + (size_t)TILE_ROWS * lstr);

    const int tid = threadIdx.x;

    // Stage weights transposed (Wt[n][k] = W[k][n]) as f16, zero-pad K..Kpad.
    for (int i = tid; i < Ncol * Kpad; i += blockDim.x) {
        int n = i / Kpad, k = i - n * Kpad;
        Wt[n * lstr + k] = (k < K) ? (_Float16)W[(size_t)k * Ncol + n] : (_Float16)0.f;
    }
    for (int n = tid; n < Ncol; n += blockDim.x) bsh[n] = bias[n];

    // Stage A tile as f16, zero-pad OOB rows / K..Kpad.
    const int row0 = blockIdx.x * TILE_ROWS;
    for (int i = tid; i < TILE_ROWS * Kpad; i += blockDim.x) {
        int r = i / Kpad, k = i - r * Kpad;
        int row = row0 + r;
        At[r * lstr + k] = (row < M && k < K) ? (_Float16)A[(long long)row * K + k]
                                              : (_Float16)0.f;
    }
    __syncthreads();

    const int wave = tid >> 5;
    const int lane = tid & 31;
    const int half = lane >> 4;     // lanes 16-31 hold the upper K / M chunks
    const int lrow = lane & 15;

    // A fragment (16x32 f16): lane = M(lrow); a[0..7] = K[kt+8*half .. +7],
    //                                          a[8..15] = K[kt+16+8*half .. +7]
    const _Float16* arow = At + (wave * 16 + lrow) * lstr;

    for (int nt = 0; nt < Ncol; nt += 16) {
        v8f acc = {};
        // B fragment (32x16 f16) from transposed weights: lane = N(lrow),
        // b[0..7] = K[kt+8*half .. +7], b[8..15] = K[kt+16+8*half .. +7] in Wt row.
        const _Float16* brow = Wt + (nt + lrow) * lstr;
        for (int kt = 0; kt < Kpad; kt += 32) {
            v8h alo = *(const v8h*)(arow + kt + 8 * half);
            v8h ahi = *(const v8h*)(arow + kt + 16 + 8 * half);
            v8h blo = *(const v8h*)(brow + kt + 8 * half);
            v8h bhi = *(const v8h*)(brow + kt + 16 + 8 * half);
            v16h a = CAT16(alo, ahi);
            v16h b = CAT16(blo, bhi);
            acc = __builtin_amdgcn_wmma_f32_16x16x32_f16(
                false, a, false, b, (short)0, acc, false, false);
        }
        // C/D layout: VGPR r -> M = r + 8*half, lane -> N = lrow.
        const int col = nt + lrow;
        const float bv = bsh[col];
        #pragma unroll
        for (int r = 0; r < 8; ++r) {
            int row = row0 + wave * 16 + r + 8 * half;
            if (row < M) {
                float v = acc[r] + bv;
                if (fuse_relu) v = fmaxf(v, 0.f);
                C[(long long)row * Ncol + col] = v;
            }
        }
    }
}

// ------------------------- ordered-float encoding --------------------------
__device__ __forceinline__ unsigned f2ord(float f) {
    unsigned u = __float_as_uint(f);
    return (u & 0x80000000u) ? ~u : (u | 0x80000000u);
}
__device__ __forceinline__ float ord2f(unsigned u) {
    return (u & 0x80000000u) ? __uint_as_float(u & 0x7fffffffu)
                             : __uint_as_float(~u);
}

// ------------------------------ fill kernels -------------------------------
__global__ void fill_f32(float* p, float v, long long n) {
    long long i = (long long)blockIdx.x * blockDim.x + threadIdx.x;
    if (i < n) p[i] = v;
}
__global__ void fill_u32(unsigned* p, unsigned v, long long n) {
    long long i = (long long)blockIdx.x * blockDim.x + threadIdx.x;
    if (i < n) p[i] = v;
}

// ------------------------------ edge kernels -------------------------------
// msg = relu(x_src[src] + e) + 1e-7 ; eM <- msg ; smax[dst] = max(t*msg)
__global__ void edge_msg_max(const float* __restrict__ xsrc,
                             float* __restrict__ eM,
                             const int* __restrict__ src,
                             const int* __restrict__ dst,
                             unsigned* __restrict__ smax,
                             const float* __restrict__ tptr,
                             long long total, int shift, int mask)
{
    long long i = (long long)blockIdx.x * blockDim.x + threadIdx.x;
    if (i >= total) return;
    int c = (int)(i & mask);
    long long e = i >> shift;
    float t = *tptr;
    float msg = fmaxf(xsrc[((long long)src[e] << shift) + c] + eM[i], 0.f) + 1e-7f;
    eM[i] = msg;
    atomicMax(&smax[((long long)dst[e] << shift) + c], f2ord(t * msg));
}

// decode ordered uint; non-finite (no in-edges) -> 0
__global__ void smax_finalize(unsigned* p, long long n) {
    long long i = (long long)blockIdx.x * blockDim.x + threadIdx.x;
    if (i >= n) return;
    float f = ord2f(p[i]);
    if (!isfinite(f)) f = 0.f;
    ((float*)p)[i] = f;
}

// den[dst] += exp(t*msg - smax[dst])
__global__ void edge_den(const float* __restrict__ eM,
                         const float* __restrict__ smax,
                         float* __restrict__ den,
                         const int* __restrict__ dst,
                         const float* __restrict__ tptr,
                         long long total, int shift, int mask)
{
    long long i = (long long)blockIdx.x * blockDim.x + threadIdx.x;
    if (i >= total) return;
    int c = (int)(i & mask);
    long long e = i >> shift;
    long long o = ((long long)dst[e] << shift) + c;
    float t = *tptr;
    atomicAdd(&den[o], __expf(t * eM[i] - smax[o]));
}

// agg[dst] += msg * exp(t*msg - smax[dst]) / max(den[dst], 1e-16)
__global__ void edge_agg(const float* __restrict__ eM,
                         const float* __restrict__ smax,
                         const float* __restrict__ den,
                         float* __restrict__ agg,
                         const int* __restrict__ dst,
                         const float* __restrict__ tptr,
                         long long total, int shift, int mask)
{
    long long i = (long long)blockIdx.x * blockDim.x + threadIdx.x;
    if (i >= total) return;
    int c = (int)(i & mask);
    long long e = i >> shift;
    long long o = ((long long)dst[e] << shift) + c;
    float t = *tptr;
    float msg = eM[i];
    float ex = __expf(t * msg - smax[o]);
    atomicAdd(&agg[o], msg * (ex / fmaxf(den[o], 1e-16f)));
}

// mix = agg + x_dst
__global__ void add_kernel(const float* a, const float* b, float* out, long long n) {
    long long i = (long long)blockIdx.x * blockDim.x + threadIdx.x;
    if (i < n) out[i] = a[i] + b[i];
}

// ------------------------------ batch norm ---------------------------------
__global__ void bn_stats(const float* __restrict__ X, int M, int Ncol,
                         float* __restrict__ mu, float* __restrict__ rstd)
{
    __shared__ float ssum[256], ssq[256];
    const int c = blockIdx.x;
    float s = 0.f, q = 0.f;
    for (int r = threadIdx.x; r < M; r += blockDim.x) {
        float v = X[(long long)r * Ncol + c];
        s += v; q += v * v;
    }
    ssum[threadIdx.x] = s; ssq[threadIdx.x] = q;
    __syncthreads();
    for (int off = 128; off > 0; off >>= 1) {
        if ((int)threadIdx.x < off) {
            ssum[threadIdx.x] += ssum[threadIdx.x + off];
            ssq[threadIdx.x]  += ssq[threadIdx.x + off];
        }
        __syncthreads();
    }
    if (threadIdx.x == 0) {
        float m   = ssum[0] / (float)M;
        float var = ssq[0] / (float)M - m * m;
        mu[c]   = m;
        rstd[c] = rsqrtf(fmaxf(var, 0.f) + 1e-5f);
    }
}

__global__ void bn_apply_relu(float* __restrict__ X,
                              const float* __restrict__ mu,
                              const float* __restrict__ rstd,
                              const float* __restrict__ gamma,
                              const float* __restrict__ beta,
                              long long total, int mask)
{
    long long i = (long long)blockIdx.x * blockDim.x + threadIdx.x;
    if (i >= total) return;
    int c = (int)(i & mask);
    X[i] = fmaxf((X[i] - mu[c]) * rstd[c] * gamma[c] + beta[c], 0.f);
}

// ------------------------------ pooling + head -----------------------------
__global__ void pool_kernel(const float* __restrict__ h,
                            const int* __restrict__ batch,
                            float* __restrict__ sums, float* __restrict__ cnt,
                            long long total)
{
    long long i = (long long)blockIdx.x * blockDim.x + threadIdx.x;
    if (i >= total) return;
    int c = (int)(i & 31);
    long long node = i >> 5;
    int g = batch[node];
    atomicAdd(&sums[g * 32 + c], h[i]);
    if (c == 0) atomicAdd(&cnt[g], 1.0f);
}

__global__ void head_kernel(const float* __restrict__ sums,
                            const float* __restrict__ cnt,
                            const float* __restrict__ fc1w, const float* __restrict__ fc1b,
                            const float* __restrict__ fc2w, const float* __restrict__ fc2b,
                            float* __restrict__ out)
{
    __shared__ float g[32];
    __shared__ float hfc[64];
    const int gi = blockIdx.x;
    if (threadIdx.x < 32)
        g[threadIdx.x] = sums[gi * 32 + threadIdx.x] / fmaxf(cnt[gi], 1.0f);
    __syncthreads();
    {
        float acc = fc1b[threadIdx.x];
        for (int k = 0; k < 32; ++k) acc += g[k] * fc1w[k * 64 + threadIdx.x];
        hfc[threadIdx.x] = fmaxf(acc, 0.f);
    }
    __syncthreads();
    if (threadIdx.x == 0) {
        float acc = fc2b[0];
        for (int k = 0; k < 64; ++k) acc += hfc[k] * fc2w[k];
        out[gi] = 1.f / (1.f + __expf(-acc));
    }
}

// ------------------------------- launcher ----------------------------------
static inline int nblk(long long n) { return (int)((n + 255) / 256); }

static inline void launch_gemm(const float* A, const float* W, const float* b,
                               float* C, int M, int K, int Ncol, int relu,
                               hipStream_t s)
{
    int Kpad = (K + 31) & ~31;
    int lstr = Kpad + 8;
    size_t sh = (size_t)(Ncol + TILE_ROWS) * lstr * sizeof(_Float16)
              + (size_t)Ncol * sizeof(float);
    wmma_gemm_bias<<<(M + TILE_ROWS - 1) / TILE_ROWS, 256, sh, s>>>(
        A, W, b, C, M, K, Ncol, relu);
}

extern "C" void kernel_launch(void* const* d_in, const int* in_sizes, int n_in,
                              void* d_out, int out_size, void* d_ws, size_t ws_size,
                              hipStream_t stream)
{
    (void)in_sizes; (void)out_size; (void)ws_size;
    if (n_in < 47) return;

    // Input order: top-level setup_inputs() insertion order; `params`
    // flattened as a JAX pytree (dict keys sorted):
    //   0: x [N,64]   1: edge_attr [E,16]
    //   2 + 13*l : conv l leaves sorted:
    //     b1, b2, beta, gamma, ld_b, ld_w, le_b, le_w, ls_b, ls_w, t, w1, w2
    //   41: fc1_b  42: fc1_w  43: fc2_b  44: fc2_w
    //   45: edge_index [2,E]  46: batch [N]
    const float* x         = (const float*)d_in[0];
    const float* edge_attr = (const float*)d_in[1];
    const float* fc1_b = (const float*)d_in[41];
    const float* fc1_w = (const float*)d_in[42];
    const float* fc2_b = (const float*)d_in[43];
    const float* fc2_w = (const float*)d_in[44];
    const int* edge_index = (const int*)d_in[45];
    const int* batch      = (const int*)d_in[46];
    const int* src = edge_index;
    const int* dst = edge_index + N_EDGES;

    // Allow the big-K GEMM tiles (up to ~136KB dynamic LDS; CDNA5 = 320KB/WGP).
    hipFuncSetAttribute((const void*)wmma_gemm_bias,
                        hipFuncAttributeMaxDynamicSharedMemorySize, 160 * 1024);

    // ----- carve workspace -----
    char* w = (char*)d_ws;
    auto carve = [&](size_t bytes) -> void* {
        void* p = (void*)w;
        w += (bytes + 255) & ~(size_t)255;
        return p;
    };
    float*    xsrc = (float*)   carve((size_t)N_NODES * 128 * 4);
    float*    xdst = (float*)   carve((size_t)N_NODES * 128 * 4);
    float*    mix  = (float*)   carve((size_t)N_NODES * 128 * 4);
    float*    hA   = (float*)   carve((size_t)N_NODES * 128 * 4);
    float*    hB   = (float*)   carve((size_t)N_NODES * 128 * 4);
    float*    hcat = (float*)   carve((size_t)N_NODES * 256 * 4);
    unsigned* smax = (unsigned*)carve((size_t)N_NODES * 128 * 4);
    float*    den  = (float*)   carve((size_t)N_NODES * 128 * 4);
    float*    agg  = (float*)   carve((size_t)N_NODES * 128 * 4);
    float*    eM   = (float*)   carve((size_t)N_EDGES * 128 * 4);
    float*    mu   = (float*)   carve(256 * 4);
    float*    rstd = (float*)   carve(256 * 4);
    float*    psum = (float*)   carve((size_t)N_GRAPH * 32 * 4);
    float*    pcnt = (float*)   carve((size_t)N_GRAPH * 4);

    const int din_[3]  = {64, 128, 64};
    const int dout_[3] = {128, 64, 32};
    float* houts[2] = {hA, hB};
    const float* h = x;

    for (int l = 0; l < 3; ++l) {
        const int base = 2 + l * 13;
        const float* b1    = (const float*)d_in[base + 0];
        const float* b2    = (const float*)d_in[base + 1];
        const float* beta  = (const float*)d_in[base + 2];
        const float* gamma = (const float*)d_in[base + 3];
        const float* ld_b  = (const float*)d_in[base + 4];
        const float* ld_w  = (const float*)d_in[base + 5];
        const float* le_b  = (const float*)d_in[base + 6];
        const float* le_w  = (const float*)d_in[base + 7];
        const float* ls_b  = (const float*)d_in[base + 8];
        const float* ls_w  = (const float*)d_in[base + 9];
        const float* tptr  = (const float*)d_in[base + 10];
        const float* w1    = (const float*)d_in[base + 11];
        const float* w2    = (const float*)d_in[base + 12];

        const int di = din_[l], dd = dout_[l];
        const int shift = (dd == 128) ? 7 : (dd == 64) ? 6 : 5;
        const int mask  = dd - 1;
        const long long nd = (long long)N_NODES * dd;
        const long long ed = (long long)N_EDGES * dd;

        // dense projections (WMMA)
        launch_gemm(h, ls_w, ls_b, xsrc, N_NODES, di, dd, 0, stream);
        launch_gemm(h, ld_w, ld_b, xdst, N_NODES, di, dd, 0, stream);
        launch_gemm(edge_attr, le_w, le_b, eM, N_EDGES, 16, dd, 0, stream);

        // softmax aggregation over dst segments
        fill_u32<<<nblk(nd), 256, 0, stream>>>(smax, 0x007FFFFFu /* ord(-inf) */, nd);
        fill_f32<<<nblk(nd), 256, 0, stream>>>(den, 0.f, nd);
        fill_f32<<<nblk(nd), 256, 0, stream>>>(agg, 0.f, nd);
        edge_msg_max<<<nblk(ed), 256, 0, stream>>>(xsrc, eM, src, dst, smax, tptr,
                                                   ed, shift, mask);
        smax_finalize<<<nblk(nd), 256, 0, stream>>>(smax, nd);
        edge_den<<<nblk(ed), 256, 0, stream>>>(eM, (const float*)smax, den, dst,
                                               tptr, ed, shift, mask);
        edge_agg<<<nblk(ed), 256, 0, stream>>>(eM, (const float*)smax, den, agg,
                                               dst, tptr, ed, shift, mask);
        add_kernel<<<nblk(nd), 256, 0, stream>>>(agg, xdst, mix, nd);

        // MLP: Linear(dd,2dd) -> BN(batch stats) -> ReLU -> Linear(2dd,dd), outer ReLU fused
        launch_gemm(mix, w1, b1, hcat, N_NODES, dd, 2 * dd, 0, stream);
        bn_stats<<<2 * dd, 256, 0, stream>>>(hcat, N_NODES, 2 * dd, mu, rstd);
        bn_apply_relu<<<nblk((long long)N_NODES * 2 * dd), 256, 0, stream>>>(
            hcat, mu, rstd, gamma, beta, (long long)N_NODES * 2 * dd, 2 * dd - 1);
        launch_gemm(hcat, w2, b2, houts[l & 1], N_NODES, 2 * dd, dd, 1, stream);
        h = houts[l & 1];
    }

    // global mean pool + head
    fill_f32<<<nblk(N_GRAPH * 32), 256, 0, stream>>>(psum, 0.f, N_GRAPH * 32);
    fill_f32<<<1, 256, 0, stream>>>(pcnt, 0.f, N_GRAPH);
    pool_kernel<<<nblk((long long)N_NODES * 32), 256, 0, stream>>>(
        h, batch, psum, pcnt, (long long)N_NODES * 32);
    head_kernel<<<N_GRAPH, 64, 0, stream>>>(psum, pcnt, fc1_w, fc1_b, fc2_w, fc2_b,
                                            (float*)d_out);
}
